// CosineSimCodebook_84035330114068
// MI455X (gfx1250) — compile-verified, matrix operand order checked by reading, not run on hardware
//
#include <hip/hip_runtime.h>

typedef _Float16 half_t;
typedef __attribute__((ext_vector_type(8)))  _Float16 v8h;
typedef __attribute__((ext_vector_type(16))) _Float16 v16h;
typedef __attribute__((ext_vector_type(8)))  float    v8f;

#define KD   512          // feature dim (fixed by reference)
#define BM   128          // rows per block
#define BN   128          // codes per chunk
#define BK   32           // k per WMMA step
#define XST  (KD + 8)     // x tile LDS stride in halves (conflict-free, 16B aligned)
#define EST  (BK + 8)     // e slice LDS stride in halves (40)
#define ESLICE (BN * EST) // halves per e-slice buffer

// ---------------- Kernel 1: inverse L2 norms of codebook rows ----------------
__global__ __launch_bounds__(256) void embed_inv_norm_kernel(
    const float* __restrict__ embed, float* __restrict__ inv_norm, int C) {
  const int lane = threadIdx.x & 31;
  const int wave = threadIdx.x >> 5;
  const int row  = blockIdx.x * 8 + wave;
  if (row >= C) return;
  const float* p = embed + (size_t)row * KD;
  float s = 0.f;
  #pragma unroll
  for (int i = 0; i < KD / 32; ++i) {
    float v = p[lane + i * 32];
    s += v * v;
  }
  #pragma unroll
  for (int m = 16; m >= 1; m >>= 1) s += __shfl_xor(s, m, 32);
  if (lane == 0) inv_norm[row] = 1.0f / fmaxf(sqrtf(s), 1e-12f);
}

// ------- Kernel 2: fused f16-WMMA GEMM + row argmax + codebook gather -------
// Waves: 4 M-groups x 2 N-groups. Each wave: 2 M-subtiles x 4 N-subtiles.
// Double-buffered e k-slice in LDS; one barrier per k-step.
__global__ __launch_bounds__(256, 1) void vq_wmma_argmax_kernel(
    const float* __restrict__ x,        // [M][KD]
    const float* __restrict__ embed,    // [C][KD]
    const float* __restrict__ inv_norm, // [C]
    float* __restrict__ out_q,          // [M][KD]
    float* __restrict__ out_ind,        // [M] (indices stored as float)
    int M, int C) {
  __shared__ alignas(16) half_t xs[BM * XST];    // persistent x tile (f16)
  __shared__ alignas(16) half_t es[2 * ESLICE];  // double-buffered e slice (f16, scaled)
  __shared__ float bval_s[BM];
  __shared__ int   bidx_s[BM];

  const int tid  = threadIdx.x;
  const int lane = tid & 31;
  const int wave = tid >> 5;
  const int hi   = lane >> 4;       // which 16-lane half
  const int ln   = lane & 15;       // m (A) or n (B/C) within a 16-subtile
  const int mgrp = wave >> 1;       // 0..3
  const int ngrp = wave & 1;        // 0..1
  const int wrow = mgrp * 32;       // this wave's 32-row strip in the block
  const int nofs = ngrp * 64;       // this wave's 64-col strip in the chunk
  const int mbase = blockIdx.x * BM;

  // ---- load & convert x tile: 128 rows x 512 f32 -> f16 LDS (coalesced) ----
  #pragma unroll 4
  for (int t = 0; t < (BM * KD / 4) / 256; ++t) {   // 64 iters of float4
    int j  = t * 256 + tid;
    int r  = j >> 7;
    int c4 = j & 127;
    const float4 v = ((const float4*)(x + (size_t)(mbase + r) * KD))[c4];
    half_t* dst = &xs[r * XST + c4 * 4];
    dst[0] = (half_t)v.x; dst[1] = (half_t)v.y;
    dst[2] = (half_t)v.z; dst[3] = (half_t)v.w;
  }

  // ---- stage e-slice for step 0 into buffer 0 ----
  {
    #pragma unroll
    for (int t = 0; t < 4; ++t) {
      int j  = t * 256 + tid;       // 0..1023 float4 over [128 rows x 8 float4]
      int r  = j >> 3;
      int c4 = j & 7;
      const float  s = inv_norm[r];
      const float4 v = ((const float4*)(embed + (size_t)r * KD))[c4];
      half_t* dst = &es[r * EST + c4 * 4];
      dst[0] = (half_t)(v.x * s); dst[1] = (half_t)(v.y * s);
      dst[2] = (half_t)(v.z * s); dst[3] = (half_t)(v.w * s);
    }
  }
  __syncthreads();

  float rbv[16];
  int   rbi[16];
  #pragma unroll
  for (int v = 0; v < 16; ++v) { rbv[v] = -3.0e38f; rbi[v] = 0; }

  v8f acc[2][4];
  #pragma unroll
  for (int mi = 0; mi < 2; ++mi)
    #pragma unroll
    for (int ni = 0; ni < 4; ++ni) { v8f z = {}; acc[mi][ni] = z; }

  const int NSTEP = (C / BN) * (KD / BK);   // flattened (chunk, kstep)

  for (int s = 0; s < NSTEP; ++s) {
    const int cur   = s & 1;
    const int kb    = (s & 15) * BK;
    const int cbase = (s >> 4) * BN;

    // ---- issue next slice's global loads early (latency hidden by WMMAs) ----
    const bool havenext = (s + 1) < NSTEP;
    float4 stg[4];
    float  sc[4];
    if (havenext) {
      const int ncb = ((s + 1) >> 4) * BN;
      const int nkb = ((s + 1) & 15) * BK;
      #pragma unroll
      for (int t = 0; t < 4; ++t) {
        int j  = t * 256 + tid;
        int r  = j >> 3;
        int c4 = j & 7;
        sc[t]  = inv_norm[ncb + r];
        stg[t] = ((const float4*)(embed + (size_t)(ncb + r) * KD + nkb))[c4];
      }
    }

    // ---- compute on current buffer: 2 A frags x 4 B frags -> 8 WMMAs ----
    const half_t* esc = es + cur * ESLICE;
    v16h afrag[2];
    #pragma unroll
    for (int mi = 0; mi < 2; ++mi) {
      const half_t* pa = &xs[(wrow + mi * 16 + ln) * XST + kb + hi * 8];
      v8h alo = *(const v8h*)pa;
      v8h ahi = *(const v8h*)(pa + 16);
      afrag[mi] = __builtin_shufflevector(alo, ahi,
          0, 1, 2, 3, 4, 5, 6, 7, 8, 9, 10, 11, 12, 13, 14, 15);
    }
    #pragma unroll
    for (int ni = 0; ni < 4; ++ni) {
      const half_t* pb = &esc[(nofs + ni * 16 + ln) * EST + hi * 8];
      v8h blo = *(const v8h*)pb;
      v8h bhi = *(const v8h*)(pb + 16);
      v16h bfrag = __builtin_shufflevector(blo, bhi,
          0, 1, 2, 3, 4, 5, 6, 7, 8, 9, 10, 11, 12, 13, 14, 15);
      #pragma unroll
      for (int mi = 0; mi < 2; ++mi)
        acc[mi][ni] = __builtin_amdgcn_wmma_f32_16x16x32_f16(
            false, afrag[mi], false, bfrag, (short)0, acc[mi][ni], false, false);
    }

    // ---- convert + store staged slice into the other buffer ----
    if (havenext) {
      half_t* dstbuf = es + ((s + 1) & 1) * ESLICE;
      #pragma unroll
      for (int t = 0; t < 4; ++t) {
        int j  = t * 256 + tid;
        int r  = j >> 3;
        int c4 = j & 7;
        half_t* dst = &dstbuf[r * EST + c4 * 4];
        dst[0] = (half_t)(stg[t].x * sc[t]); dst[1] = (half_t)(stg[t].y * sc[t]);
        dst[2] = (half_t)(stg[t].z * sc[t]); dst[3] = (half_t)(stg[t].w * sc[t]);
      }
    }

    // ---- end of a 128-col chunk: fused argmax update, reset accumulators ----
    if ((s & 15) == 15) {
      #pragma unroll
      for (int mi = 0; mi < 2; ++mi) {
        #pragma unroll
        for (int v = 0; v < 8; ++v) {
          // C/D layout: element v, lane -> row = v + 8*hi, col = nsub*16 + ln
          float bv = acc[mi][0][v];
          int   bi = cbase + nofs + ln;
          #pragma unroll
          for (int ni = 1; ni < 4; ++ni) {    // ascending idx: '>' keeps first max
            float cv = acc[mi][ni][v];
            int   ci = cbase + nofs + ni * 16 + ln;
            if (cv > bv) { bv = cv; bi = ci; }
          }
          #pragma unroll
          for (int m = 1; m <= 8; m <<= 1) {  // reduce the 16 lanes of this row
            float ov = __shfl_xor(bv, m, 32);
            int   oi = __shfl_xor(bi, m, 32);
            if (ov > bv || (ov == bv && oi < bi)) { bv = ov; bi = oi; }
          }
          const int q = mi * 8 + v;
          if (bv > rbv[q]) { rbv[q] = bv; rbi[q] = bi; }  // earlier chunk wins ties
        }
      }
      #pragma unroll
      for (int mi = 0; mi < 2; ++mi)
        #pragma unroll
        for (int ni = 0; ni < 4; ++ni) { v8f z = {}; acc[mi][ni] = z; }
    }
    __syncthreads();   // single barrier per k-step (double buffer)
  }

  // ---- combine the two column-half winners per row (once per block) ----
  if (ngrp == 0 && ln == 0) {
    #pragma unroll
    for (int mi = 0; mi < 2; ++mi)
      #pragma unroll
      for (int v = 0; v < 8; ++v) {
        int r = wrow + mi * 16 + hi * 8 + v;
        bval_s[r] = rbv[mi * 8 + v];
        bidx_s[r] = rbi[mi * 8 + v];
      }
  }
  __syncthreads();
  if (ngrp == 1 && ln == 0) {
    #pragma unroll
    for (int mi = 0; mi < 2; ++mi)
      #pragma unroll
      for (int v = 0; v < 8; ++v) {
        int r = wrow + mi * 16 + hi * 8 + v;
        float mv = rbv[mi * 8 + v];
        int   ix = rbi[mi * 8 + v];
        if (mv > bval_s[r] || (mv == bval_s[r] && ix < bidx_s[r])) {
          bval_s[r] = mv; bidx_s[r] = ix;
        }
      }
  }
  __syncthreads();

  if (tid < BM) out_ind[mbase + tid] = (float)bidx_s[tid];

  // ---- gather quantize = embed[best_idx] (unnormalized), coalesced float4 ----
  #pragma unroll 4
  for (int t = 0; t < (BM * KD / 4) / 256; ++t) {
    int j  = t * 256 + tid;
    int r  = j >> 7;
    int c4 = j & 127;
    const float4 v = ((const float4*)(embed + (size_t)bidx_s[r] * KD))[c4];
    ((float4*)(out_q + (size_t)(mbase + r) * KD))[c4] = v;
  }
}

extern "C" void kernel_launch(void* const* d_in, const int* in_sizes, int n_in,
                              void* d_out, int out_size, void* d_ws, size_t ws_size,
                              hipStream_t stream) {
  const float* x     = (const float*)d_in[0];   // [8,1024,512]
  const float* embed = (const float*)d_in[1];   // [1,16384,512]
  const int M = in_sizes[0] / KD;               // 8192
  const int C = in_sizes[1] / KD;               // 16384

  float* inv_norm = (float*)d_ws;               // C floats (64 KB scratch)
  float* out_q    = (float*)d_out;              // [M][KD]
  float* out_ind  = out_q + (size_t)M * KD;     // [M]

  embed_inv_norm_kernel<<<(C + 7) / 8, 256, 0, stream>>>(embed, inv_norm, C);
  vq_wmma_argmax_kernel<<<M / BM, 256, 0, stream>>>(x, embed, inv_norm,
                                                    out_q, out_ind, M, C);
}
